// PatchCore_27874337751106
// MI455X (gfx1250) — compile-verified, matrix operand order checked by reading, not run on hardware
//
#include <hip/hip_runtime.h>

// ---------------------------------------------------------------------------
// PatchCore pipeline for MI455X (gfx1250, wave32, WMMA)
// ---------------------------------------------------------------------------

typedef __attribute__((ext_vector_type(16))) __bf16        v16bf;
typedef __attribute__((ext_vector_type(8)))  float         v8f;
typedef __attribute__((ext_vector_type(4)))  unsigned int  u32x4;

#define BATCH    8
#define GRID_    24
#define CDIM     768
#define NPATCH   576                 // 24*24
#define NROWS    (BATCH * NPATCH)    // 4608 = 36*128
#define TGT      1024
#define NMEM     20000
#define NMEM_PAD 20096               // 157*128
#define IMG      384
#define LDS_STRIDE 40                // bf16 elems per LDS row (80B, 16B aligned, conflict-free)

__device__ __forceinline__ unsigned short f2bf(float f) {
    unsigned u = __float_as_uint(f);
    unsigned r = u + 0x7FFFu + ((u >> 16) & 1u);   // round-to-nearest-even
    return (unsigned short)(r >> 16);
}
// monotone float -> uint key (unsigned atomicMin == float min, handles negatives)
__device__ __forceinline__ unsigned enc_key(float f) {
    unsigned u = __float_as_uint(f);
    return (u & 0x80000000u) ? ~u : (u | 0x80000000u);
}
__device__ __forceinline__ float dec_key(unsigned k) {
    unsigned u = (k & 0x80000000u) ? (k ^ 0x80000000u) : ~k;
    return __uint_as_float(u);
}

// ---------------------------------------------------------------------------
// keys init
// ---------------------------------------------------------------------------
__global__ void init_keys(unsigned* __restrict__ keys) {
    int i = blockIdx.x * 256 + threadIdx.x;
    if (i < NROWS) keys[i] = 0xFFFFFFFFu;
}

// ---------------------------------------------------------------------------
// Fused unfold3x3 -> Mpre pool -> Magg pool.  emb[n,t] for n=b*576+h*24+w.
// Mpre window for idx i (lin=6912,lout=1024): s=(27i)>>2, e=(27(i+1)+3)>>2.
// Magg (3072->1024) is exact mean of 3 consecutive, spanning the layer concat.
// One block per n; also emits bf16(emb) and ||emb_n||^2.
// ---------------------------------------------------------------------------
__global__ void __launch_bounds__(256) emb_kernel(
        const float* __restrict__ f6, const float* __restrict__ f8,
        const float* __restrict__ f10,
        unsigned short* __restrict__ embB, float* __restrict__ en2) {
    int n = blockIdx.x;
    int b = n / NPATCH, p = n - b * NPATCH;
    int h = p / GRID_,  w = p - h * GRID_;
    int tid = threadIdx.x;
    float sq = 0.f;

    for (int t = tid; t < TGT; t += 256) {
        float acc = 0.f;
        #pragma unroll
        for (int m3 = 0; m3 < 3; ++m3) {
            int j = 3 * t + m3;
            int l = j >> 10;                  // source layer
            int i = j & 1023;                 // index within layer's 1024
            int s = (27 * i) >> 2;
            int e = (27 * (i + 1) + 3) >> 2;
            const float* F = (l == 0) ? f6 : ((l == 1) ? f8 : f10);
            F += (size_t)b * (NPATCH + 1) * CDIM;
            float sum = 0.f;
            for (int k = s; k < e; ++k) {     // 7..8 elems of the 6912 unfold vec
                int c  = k / 9;
                int r  = k - 9 * c;
                int kh = r / 3;
                int kw = r - 3 * kh;
                int hh = h + kh - 1, ww = w + kw - 1;
                float v = 0.f;
                if ((unsigned)hh < (unsigned)GRID_ && (unsigned)ww < (unsigned)GRID_)
                    v = F[(size_t)(1 + hh * GRID_ + ww) * CDIM + c];
                sum += v;
            }
            acc += sum / (float)(e - s);
        }
        float v = acc * (1.0f / 3.0f);
        embB[(size_t)n * TGT + t] = f2bf(v);
        sq += v * v;
    }
    __shared__ float red[256];
    red[tid] = sq; __syncthreads();
    for (int s = 128; s > 0; s >>= 1) { if (tid < s) red[tid] += red[tid + s]; __syncthreads(); }
    if (tid == 0) en2[n] = red[0];
}

// ---------------------------------------------------------------------------
// memory bank -> bf16 (padded rows zeroed) + row squared norms (pad = +huge)
// ---------------------------------------------------------------------------
__global__ void __launch_bounds__(256) memprep(
        const float* __restrict__ mb,
        unsigned short* __restrict__ memB, float* __restrict__ mn2) {
    int m = blockIdx.x, tid = threadIdx.x;
    float sq = 0.f;
    if (m < NMEM) {
        for (int k = tid; k < TGT; k += 256) {
            float v = mb[(size_t)m * TGT + k];
            memB[(size_t)m * TGT + k] = f2bf(v);
            sq += v * v;
        }
    } else {
        for (int k = tid; k < TGT; k += 256) memB[(size_t)m * TGT + k] = 0;
    }
    __shared__ float red[256];
    red[tid] = sq; __syncthreads();
    for (int s = 128; s > 0; s >>= 1) { if (tid < s) red[tid] += red[tid + s]; __syncthreads(); }
    if (tid == 0) mn2[m] = (m < NMEM) ? red[0] : 3.0e38f;
}

// ---------------------------------------------------------------------------
// WMMA distance GEMM + fused min.  Grid (36, 157); WG = 256 thr = 8 waves.
// Tile 128(n) x 128(m), K step 32 (v_wmma_f32_16x16x32_bf16).
// Wave w: wn=w&3 -> 32 emb rows, wm=w>>2 -> 64 bank cols; 2x4 wmma tiles.
// ---------------------------------------------------------------------------
__global__ void __launch_bounds__(256, 2) nn_gemm(
        const unsigned short* __restrict__ embB,
        const unsigned short* __restrict__ memB,
        const float* __restrict__ en2, const float* __restrict__ mn2,
        unsigned* __restrict__ keys) {
    __shared__ __align__(16) unsigned short sA[128 * LDS_STRIDE];
    __shared__ __align__(16) unsigned short sB[128 * LDS_STRIDE];
    __shared__ unsigned sKeys[128];

    const int tid  = threadIdx.x;
    const int n0   = blockIdx.x * 128;
    const int m0   = blockIdx.y * 128;
    const int wv   = tid >> 5, lane = tid & 31;
    const int wn   = wv & 3,   wm   = wv >> 2;
    const int mrow = lane & 15, half = lane >> 4;
    const int lrow = tid >> 2,  lcg  = tid & 3;   // global->LDS: 4 thr/row, 16B each

    if (tid < 128) sKeys[tid] = 0xFFFFFFFFu;

    v8f acc[2][4] = {};

    for (int kt = 0; kt < TGT; kt += 32) {
        __syncthreads();
        #pragma unroll
        for (int q = 0; q < 2; ++q) {
            int row = lrow + q * 64;
            u32x4 a  = *(const u32x4*)(embB + (size_t)(n0 + row) * TGT + kt + lcg * 8);
            u32x4 bb = *(const u32x4*)(memB + (size_t)(m0 + row) * TGT + kt + lcg * 8);
            *(u32x4*)&sA[row * LDS_STRIDE + lcg * 8] = a;
            *(u32x4*)&sB[row * LDS_STRIDE + lcg * 8] = bb;
        }
        __syncthreads();

        union FragU { u32x4 q[2]; v16bf v; };
        FragU fa[2], fb[4];
        // CDNA5 16-bit A layout: lane=row, lane-half selects K {0..7,16..23} vs {8..15,24..31}
        #pragma unroll
        for (int a = 0; a < 2; ++a) {
            int r = wn * 32 + a * 16 + mrow;
            fa[a].q[0] = *(const u32x4*)&sA[r * LDS_STRIDE + half * 8];
            fa[a].q[1] = *(const u32x4*)&sA[r * LDS_STRIDE + 16 + half * 8];
        }
        #pragma unroll
        for (int b = 0; b < 4; ++b) {
            int r = wm * 64 + b * 16 + mrow;
            fb[b].q[0] = *(const u32x4*)&sB[r * LDS_STRIDE + half * 8];
            fb[b].q[1] = *(const u32x4*)&sB[r * LDS_STRIDE + 16 + half * 8];
        }
        #pragma unroll
        for (int a = 0; a < 2; ++a)
            #pragma unroll
            for (int b = 0; b < 4; ++b)
                acc[a][b] = __builtin_amdgcn_wmma_f32_16x16x32_bf16(
                    false, fa[a].v, false, fb[b].v, (short)0, acc[a][b], false, false);
    }
    __syncthreads();

    // d^2 = |e|^2 + |m|^2 - 2 e.m ; min over this block's 128 bank rows.
    float m2[4];
    #pragma unroll
    for (int b = 0; b < 4; ++b) m2[b] = mn2[m0 + wm * 64 + b * 16 + mrow];
    #pragma unroll
    for (int a = 0; a < 2; ++a) {
        #pragma unroll
        for (int r = 0; r < 8; ++r) {
            int nl = wn * 32 + a * 16 + half * 8 + r;   // C layout: lane-half -> M+8
            float e2 = en2[n0 + nl];
            float best = 3.0e38f;
            #pragma unroll
            for (int b = 0; b < 4; ++b) {
                float d = e2 + m2[b] - 2.0f * acc[a][b][r];
                best = fminf(best, d);
            }
            atomicMin(&sKeys[nl], enc_key(best));
        }
    }
    __syncthreads();
    if (tid < 128) atomicMin(&keys[n0 + tid], sKeys[tid]);
}

// ---------------------------------------------------------------------------
// image_scores[b] = max over 576 patch scores
// ---------------------------------------------------------------------------
__global__ void __launch_bounds__(256) score_kernel(
        const unsigned* __restrict__ keys, float* __restrict__ out) {
    int b = blockIdx.x, tid = threadIdx.x;
    float best = -3.0e38f;
    for (int i = tid; i < NPATCH; i += 256) best = fmaxf(best, dec_key(keys[b * NPATCH + i]));
    __shared__ float red[256];
    red[tid] = best; __syncthreads();
    for (int s = 128; s > 0; s >>= 1) { if (tid < s) red[tid] = fmaxf(red[tid], red[tid + s]); __syncthreads(); }
    if (tid == 0) out[b] = red[0];
}

// ---------------------------------------------------------------------------
// bilinear 24x24 -> 384x384 (half-pixel centers, edge clamp)
// ---------------------------------------------------------------------------
__global__ void __launch_bounds__(256) mask_kernel(
        const unsigned* __restrict__ keys, float* __restrict__ out) {
    int idx = blockIdx.x * 256 + threadIdx.x;
    if (idx >= BATCH * IMG * IMG) return;
    int b   = idx / (IMG * IMG);
    int rem = idx - b * IMG * IMG;
    int y   = rem / IMG, x = rem - y * IMG;
    const float scale = (float)GRID_ / (float)IMG;   // 1/16
    float fy = (y + 0.5f) * scale - 0.5f;
    float fx = (x + 0.5f) * scale - 0.5f;
    float yf = floorf(fy), xf = floorf(fx);
    float wy = fy - yf,    wx = fx - xf;
    int y0 = (int)yf, x0 = (int)xf;
    int iy0 = min(max(y0, 0), GRID_ - 1), iy1 = min(max(y0 + 1, 0), GRID_ - 1);
    int ix0 = min(max(x0, 0), GRID_ - 1), ix1 = min(max(x0 + 1, 0), GRID_ - 1);
    const unsigned* kb = keys + b * NPATCH;
    float v00 = dec_key(kb[iy0 * GRID_ + ix0]);
    float v01 = dec_key(kb[iy0 * GRID_ + ix1]);
    float v10 = dec_key(kb[iy1 * GRID_ + ix0]);
    float v11 = dec_key(kb[iy1 * GRID_ + ix1]);
    float v = (1.f - wy) * ((1.f - wx) * v00 + wx * v01)
            +        wy  * ((1.f - wx) * v10 + wx * v11);
    out[8 + idx] = v;
}

// ---------------------------------------------------------------------------
extern "C" void kernel_launch(void* const* d_in, const int* in_sizes, int n_in,
                              void* d_out, int out_size, void* d_ws, size_t ws_size,
                              hipStream_t stream) {
    (void)in_sizes; (void)n_in; (void)out_size; (void)ws_size;
    const float* f6  = (const float*)d_in[0];
    const float* f8  = (const float*)d_in[1];
    const float* f10 = (const float*)d_in[2];
    const float* mb  = (const float*)d_in[3];
    float* out = (float*)d_out;

    char* ws = (char*)d_ws;
    size_t off = 0;
    unsigned short* embB = (unsigned short*)(ws + off); off += (size_t)NROWS * TGT * 2;
    float*          en2  = (float*)(ws + off);          off += (size_t)NROWS * 4;
    off = (off + 255) & ~(size_t)255;
    unsigned short* memB = (unsigned short*)(ws + off); off += (size_t)NMEM_PAD * TGT * 2;
    float*          mn2  = (float*)(ws + off);          off += (size_t)NMEM_PAD * 4;
    unsigned*       keys = (unsigned*)(ws + off);       off += (size_t)NROWS * 4;

    init_keys<<<(NROWS + 255) / 256, 256, 0, stream>>>(keys);
    emb_kernel<<<NROWS, 256, 0, stream>>>(f6, f8, f10, embB, en2);
    memprep<<<NMEM_PAD, 256, 0, stream>>>(mb, memB, mn2);
    nn_gemm<<<dim3(NROWS / 128, NMEM_PAD / 128), 256, 0, stream>>>(embB, memB, en2, mn2, keys);
    score_kernel<<<BATCH, 256, 0, stream>>>(keys, out);
    mask_kernel<<<(BATCH * IMG * IMG + 255) / 256, 256, 0, stream>>>(keys, out);
}
